// RelativeSelfAttention_87625922773572
// MI455X (gfx1250) — matrix operand order repaired and checked by reading, not compiled, not run on hardware
//
#include <hip/hip_runtime.h>
#include <hip/hip_bf16.h>

// Problem constants (match reference)
#define NBATCH 2
#define SEQ    2048
#define HDIM   1024
#define NHEADS 16
#define HEADD  64
#define MAXREL 4

#define KPITCH 72   // sK row pitch (halves): 64 + 8 pad, 144B (16B-aligned)
#define VPITCH 40   // sV/sP row pitch (halves): 32 + 8 pad, 80B (16B-aligned)

typedef __attribute__((ext_vector_type(16))) _Float16 v16h;
typedef __attribute__((ext_vector_type(8)))  _Float16 v8h;
typedef __attribute__((ext_vector_type(8)))  float    v8f;

// ---------------------------------------------------------------------------
// WMMA helper (CDNA5 wave32, V_WMMA_F32_16X16X32_F16)
// ---------------------------------------------------------------------------
__device__ __forceinline__ v8f wmma_f16(v16h a, v16h b, v8f c) {
  return __builtin_amdgcn_wmma_f32_16x16x32_f16(false, a, false, b, (short)0, c,
                                                false, false);
}

// Async global->LDS copy of 16B per lane (GLOBAL_LOAD_ASYNC_TO_LDS_B128,
// ASYNCcnt-tracked).  ldsOff = low 32 bits of the generic pointer (flat LDS
// aperture keeps the LDS offset in addr[31:0]).
__device__ __forceinline__ void async_copy_b128(const void* gsrc, void* ldst) {
  const unsigned long long ga = (unsigned long long)(uintptr_t)gsrc;
  const unsigned int       la = (unsigned int)(uintptr_t)ldst;
  asm volatile("global_load_async_to_lds_b128 %0, %1, off"
               :: "v"(la), "v"(ga) : "memory");
}
__device__ __forceinline__ void async_wait0() {
  asm volatile("s_wait_asynccnt 0" ::: "memory");
}

// 16x32 A-tile (M x K) from row-major f16 matrix.
// ISA layout: lanes 0-15: M=lane, elems 0-7 -> K=0..7, elems 8-15 -> K=16..23;
//             lanes 16-31 shift K by 8 / 24.  Two aligned b128 loads.
__device__ __forceinline__ v16h load_a_h(const _Float16* base, int ld,
                                         int row0, int k0, int lane) {
  const int m  = lane & 15;
  const int kh = (lane >> 4) & 1;
  const _Float16* p = base + (size_t)(row0 + m) * ld + k0 + kh * 8;
  v8h lo = *(const v8h*)(p);
  v8h hi = *(const v8h*)(p + 16);
  return __builtin_shufflevector(lo, hi, 0, 1, 2, 3, 4, 5, 6, 7,
                                 8, 9, 10, 11, 12, 13, 14, 15);
}

// 32x16 B-tile (K x N), B[k][n] = src[(n0+n)*ld + k0+k]  (K contiguous).
// ISA layout: lanes 0-15: N=lane, elems 0-7 -> K=0..7, elems 8-15 -> K=16..23;
//             lanes 16-31 shift K by 8 / 24.
__device__ __forceinline__ v16h load_bT_h(const _Float16* base, int ld,
                                          int k0, int n0, int lane) {
  const int n  = lane & 15;
  const int kb = k0 + ((lane >> 4) & 1) * 8;
  const _Float16* p = base + (size_t)(n0 + n) * ld + kb;
  v8h lo = *(const v8h*)(p);
  v8h hi = *(const v8h*)(p + 16);
  return __builtin_shufflevector(lo, hi, 0, 1, 2, 3, 4, 5, 6, 7,
                                 8, 9, 10, 11, 12, 13, 14, 15);
}

// ---------------------------------------------------------------------------
// Kernel 0: one-shot f32 -> f16 conversion (x and the four weight matrices).
// ---------------------------------------------------------------------------
__global__ void cvt_f16_kernel(const float* __restrict__ src,
                               _Float16* __restrict__ dst, int n) {
  const int i = (blockIdx.x * 256 + threadIdx.x) * 8;
  if (i >= n) return;
  const float4 a = *(const float4*)(src + i);
  const float4 b = *(const float4*)(src + i + 4);
  v8h o;
  o[0] = (_Float16)a.x; o[1] = (_Float16)a.y;
  o[2] = (_Float16)a.z; o[3] = (_Float16)a.w;
  o[4] = (_Float16)b.x; o[5] = (_Float16)b.y;
  o[6] = (_Float16)b.z; o[7] = (_Float16)b.w;
  *(v8h*)(dst + i) = o;
}

// ---------------------------------------------------------------------------
// Kernel 1: fused QKV projection, 32x32 output tile per wave.
// q,k f16 row-major [b,h,t,d]; v f16 transposed [b,h,d,t].
// ---------------------------------------------------------------------------
__global__ void qkv_kernel(const _Float16* __restrict__ xh,
                           const _Float16* __restrict__ wh,   // Wq|Wk|Wv|Wo
                           const float* __restrict__ bq, const float* __restrict__ bk,
                           const float* __restrict__ bv,
                           _Float16* __restrict__ qo, _Float16* __restrict__ ko,
                           _Float16* __restrict__ vo) {
  const int lane = threadIdx.x & 31;
  const int wid  = threadIdx.x >> 5;
  const int id   = blockIdx.x * 8 + wid;       // 12288 tiles (128 x 96)
  const int tileR = id / 96;
  const int tileC = id % 96;
  const int seg = tileC >> 5;                  // 0=Q, 1=K, 2=V
  const int oc0 = (tileC & 31) * 32;
  const _Float16* W = wh + (size_t)seg * HDIM * HDIM;
  const float* bias = (seg == 0) ? bq : (seg == 1) ? bk : bv;
  const int row0 = tileR * 32;

  v8f c00 = {}, c01 = {}, c10 = {}, c11 = {};
#pragma unroll 2
  for (int k0 = 0; k0 < HDIM; k0 += 32) {
    const v16h a0 = load_a_h(xh, HDIM, row0,      k0, lane);
    const v16h a1 = load_a_h(xh, HDIM, row0 + 16, k0, lane);
    const v16h b0 = load_bT_h(W, HDIM, k0, oc0,      lane);
    const v16h b1 = load_bT_h(W, HDIM, k0, oc0 + 16, lane);
    c00 = wmma_f16(a0, b0, c00);
    c01 = wmma_f16(a0, b1, c01);
    c10 = wmma_f16(a1, b0, c10);
    c11 = wmma_f16(a1, b1, c11);
  }

  const int n  = lane & 15;
  const int hi = (lane >> 4) & 1;
#pragma unroll
  for (int ct = 0; ct < 4; ++ct) {
    const v8f c = (ct == 0) ? c00 : (ct == 1) ? c01 : (ct == 2) ? c10 : c11;
    const int rbase = (ct >> 1) * 16;
    const int col   = oc0 + (ct & 1) * 16 + n;   // within segment [0,1024)
    const float bb  = bias[col];
    const int head  = col >> 6;
    const int d     = col & 63;
#pragma unroll
    for (int r = 0; r < 8; ++r) {
      const int g  = row0 + rbase + r + hi * 8;  // flattened (b,t)
      const int bI = g >> 11;
      const int t  = g & (SEQ - 1);
      const int bh = bI * NHEADS + head;
      const _Float16 val = (_Float16)(c[r] + bb);
      if (seg == 0)      qo[((size_t)bh * SEQ + t) * HEADD + d] = val;
      else if (seg == 1) ko[((size_t)bh * SEQ + t) * HEADD + d] = val;
      else               vo[((size_t)bh * HEADD + d) * SEQ + t] = val;
    }
  }
}

// ---------------------------------------------------------------------------
// Kernel 2: flash attention. Block = 8 waves sharing one (b,head); each wave
// owns a 16-query tile.  Per 32-key step the block stages the K-tile (32x64)
// and V-tile (64x32) into double-buffered LDS with ASYNC global->LDS copies
// (one b128 per thread each), issued before the compute phase of tile kt and
// waited with s_wait_asynccnt just before the end-of-iteration barrier
// (barriers do NOT cover ASYNCcnt).  Softmax denominator is computed on the
// matrix pipe: l = wmma(P, ones, alpha*l).
// ---------------------------------------------------------------------------
__global__ void attn_kernel(const _Float16* __restrict__ q,
                            const _Float16* __restrict__ k,
                            const _Float16* __restrict__ vT,
                            const int* __restrict__ mask,
                            const float* __restrict__ rel_bias,
                            _Float16* __restrict__ hid) {
  __shared__ _Float16 sK[2][32 * KPITCH];      // 2 x 4.5 KB, rows = keys
  __shared__ _Float16 sV[2][64 * VPITCH];      // 2 x 5   KB, rows = d
  __shared__ _Float16 sP[8][16 * VPITCH];      // 10 KB, per-wave P staging
  const int tid  = threadIdx.x;
  const int lane = tid & 31;
  const int wid  = tid >> 5;

  const int blk = blockIdx.x;                  // 512 blocks
  const int bh  = blk >> 4;                    // 16 blocks per (b,head)
  const int qt  = (blk & 15) * 8 + wid;        // this wave's query tile
  const int b   = bh >> 4;
  const int h   = bh & 15;

  const _Float16* qb = q  + (size_t)bh * SEQ * HEADD;
  const _Float16* kb = k  + (size_t)bh * SEQ * HEADD;
  const _Float16* vb = vT + (size_t)bh * HEADD * SEQ;
  const int*   maskb = mask + b * SEQ;
  const float* rbh   = rel_bias + h;

  const int n  = lane & 15;
  const int hi = (lane >> 4) & 1;

  // Cooperative staging geometry: one b128 per thread per tile.
  const int kRow = tid >> 3, kCol = (tid & 7) * 8;   // K tile: 32 x 64
  const int vRow = tid >> 2, vCol = (tid & 3) * 8;   // V tile: 64 x 32
  const _Float16* kSrc = kb + (size_t)kRow * HEADD + kCol;
  const _Float16* vSrc = vb + (size_t)vRow * SEQ + vCol;
  const int kDst = kRow * KPITCH + kCol;
  const int vDst = vRow * VPITCH + vCol;

  const v16h aQ0 = load_a_h(qb, HEADD, qt * 16, 0,  lane);
  const v16h aQ1 = load_a_h(qb, HEADD, qt * 16, 32, lane);

  v16h ones;
#pragma unroll
  for (int i = 0; i < 16; ++i) ones[i] = (_Float16)1.0f;

  v8f O0 = {}, O1 = {}, O2 = {}, O3 = {}, Lacc = {};
  float m_[8];
#pragma unroll
  for (int r = 0; r < 8; ++r) m_[r] = -1e30f;

  // Preload tile 0 into buffer 0 (async).
  async_copy_b128(kSrc, &sK[0][kDst]);
  async_copy_b128(vSrc, &sV[0][vDst]);
  async_wait0();
  __syncthreads();

  for (int kt = 0; kt < SEQ / 32; ++kt) {
    const int key0 = kt * 32;
    const int buf  = kt & 1;
    const bool pre = (kt + 1) < (SEQ / 32);
    if (pre) {                                  // kick off next tile's DMA
      async_copy_b128(kSrc + (size_t)(kt + 1) * 32 * HEADD, &sK[buf ^ 1][kDst]);
      async_copy_b128(vSrc + (kt + 1) * 32,                 &sV[buf ^ 1][vDst]);
    }

    // ---- S = Q K^T over this key tile (B-operand from LDS) ----
    v8f S0 = {}, S1 = {};
    {
      v16h bK = load_bT_h(&sK[buf][0], KPITCH, 0,  0,  lane);
      S0 = wmma_f16(aQ0, bK, S0);
      bK = load_bT_h(&sK[buf][0], KPITCH, 32, 0,  lane);
      S0 = wmma_f16(aQ1, bK, S0);
      bK = load_bT_h(&sK[buf][0], KPITCH, 0,  16, lane);
      S1 = wmma_f16(aQ0, bK, S1);
      bK = load_bT_h(&sK[buf][0], KPITCH, 32, 16, lane);
      S1 = wmma_f16(aQ1, bK, S1);
    }
    const int key_a = key0 + n;
    const int key_b = key0 + 16 + n;
    const int mk_a  = maskb[key_a];
    const int mk_b  = maskb[key_b];

    float s0[8], s1[8];
#pragma unroll
    for (int r = 0; r < 8; ++r) {
      const int tq = qt * 16 + r + hi * 8;
      int da = key_a - tq; da = min(max(da, -MAXREL), MAXREL);
      int db = key_b - tq; db = min(max(db, -MAXREL), MAXREL);
      s0[r] = mk_a ? (S0[r] * 0.125f + rbh[(da + MAXREL) * NHEADS]) : -10000.f;
      s1[r] = mk_b ? (S1[r] * 0.125f + rbh[(db + MAXREL) * NHEADS]) : -10000.f;
    }
    // ---- online softmax (row M = r + 8*hi spans 16 lanes of one half) ----
#pragma unroll
    for (int r = 0; r < 8; ++r) {
      float rm = fmaxf(s0[r], s1[r]);
#pragma unroll
      for (int off = 1; off < 16; off <<= 1) rm = fmaxf(rm, __shfl_xor(rm, off));
      const float nm = fmaxf(m_[r], rm);
      const float al = __expf(m_[r] - nm);
      const float p0 = __expf(s0[r] - nm);
      const float p1 = __expf(s1[r] - nm);
      m_[r] = nm;
      O0[r] *= al; O1[r] *= al; O2[r] *= al; O3[r] *= al; Lacc[r] *= al;
      const int M = r + hi * 8;
      sP[wid][M * VPITCH + n]      = (_Float16)p0;
      sP[wid][M * VPITCH + 16 + n] = (_Float16)p1;
    }
    // ---- O += P V, l += P 1 (P via per-wave LDS transpose) ----
    {
      const _Float16* pp = &sP[wid][n * VPITCH + hi * 8];
      v8h plo = *(const v8h*)(pp);
      v8h phi = *(const v8h*)(pp + 16);
      v16h aP = __builtin_shufflevector(plo, phi, 0, 1, 2, 3, 4, 5, 6, 7,
                                        8, 9, 10, 11, 12, 13, 14, 15);
      v16h bV = load_bT_h(&sV[buf][0], VPITCH, 0, 0,  lane);
      O0 = wmma_f16(aP, bV, O0);
      bV = load_bT_h(&sV[buf][0], VPITCH, 0, 16, lane);
      O1 = wmma_f16(aP, bV, O1);
      bV = load_bT_h(&sV[buf][0], VPITCH, 0, 32, lane);
      O2 = wmma_f16(aP, bV, O2);
      bV = load_bT_h(&sV[buf][0], VPITCH, 0, 48, lane);
      O3 = wmma_f16(aP, bV, O3);
      Lacc = wmma_f16(aP, ones, Lacc);         // row-sums on the matrix pipe
    }
    if (pre) async_wait0();   // ASYNC writes NOT covered by the barrier
    __syncthreads();
  }
  // Normalize; store hidden f16 in [b, t, head*64 + d] layout.
#pragma unroll
  for (int r = 0; r < 8; ++r) {
    const int tq = qt * 16 + r + hi * 8;
    const float inv = 1.0f / Lacc[r];
    const size_t rowoff = ((size_t)b * SEQ + tq) * HDIM + h * HEADD;
    hid[rowoff + 0  + n] = (_Float16)(O0[r] * inv);
    hid[rowoff + 16 + n] = (_Float16)(O1[r] * inv);
    hid[rowoff + 32 + n] = (_Float16)(O2[r] * inv);
    hid[rowoff + 48 + n] = (_Float16)(O3[r] * inv);
  }
}

// ---------------------------------------------------------------------------
// Kernel 3: output projection (32x32 tiles) + query-side mask, f32 output.
// ---------------------------------------------------------------------------
__global__ void outproj_kernel(const _Float16* __restrict__ hidh,
                               const _Float16* __restrict__ woh,
                               const float* __restrict__ bo,
                               const int* __restrict__ mask,
                               float* __restrict__ out) {
  const int lane = threadIdx.x & 31;
  const int wid  = threadIdx.x >> 5;
  const int id   = blockIdx.x * 8 + wid;       // 4096 tiles (128 x 32)
  const int tileR = id >> 5;
  const int tileC = id & 31;
  const int row0 = tileR * 32;
  const int oc0  = tileC * 32;

  v8f c00 = {}, c01 = {}, c10 = {}, c11 = {};
#pragma unroll 2
  for (int k0 = 0; k0 < HDIM; k0 += 32) {
    const v16h a0 = load_a_h(hidh, HDIM, row0,      k0, lane);
    const v16h a1 = load_a_h(hidh, HDIM, row0 + 16, k0, lane);
    const v16h b0 = load_bT_h(woh, HDIM, k0, oc0,      lane);
    const v16h b1 = load_bT_h(woh, HDIM, k0, oc0 + 16, lane);
    c00 = wmma_f16(a0, b0, c00);
    c01 = wmma_f16(a0, b1, c01);
    c10 = wmma_f16(a1, b0, c10);
    c11 = wmma_f16(a1, b1, c11);
  }

  const int n  = lane & 15;
  const int hi = (lane >> 4) & 1;
#pragma unroll
  for (int ct = 0; ct < 4; ++ct) {
    const v8f c = (ct == 0) ? c00 : (ct == 1) ? c01 : (ct == 2) ? c10 : c11;
    const int rbase = (ct >> 1) * 16;
    const int col   = oc0 + (ct & 1) * 16 + n;
    const float bb  = bo[col];
#pragma unroll
    for (int r = 0; r < 8; ++r) {
      const int g = row0 + rbase + r + hi * 8;   // flattened (b,t)
      const float mk = (float)mask[g];
      out[(size_t)g * HDIM + col] = (c[r] + bb) * mk;
    }
  }
}

// ---------------------------------------------------------------------------
extern "C" void kernel_launch(void* const* d_in, const int* in_sizes, int n_in,
                              void* d_out, int out_size, void* d_ws, size_t ws_size,
                              hipStream_t stream) {
  const float* x         = (const float*)d_in[0];
  const int*   text_mask = (const int*)  d_in[1];
  const float* Wq        = (const float*)d_in[2];
  const float* bq        = (const float*)d_in[3];
  const float* Wk        = (const float*)d_in[4];
  const float* bk        = (const float*)d_in[5];
  const float* Wv        = (const float*)d_in[6];
  const float* bv        = (const float*)d_in[7];
  const float* Wo        = (const float*)d_in[8];
  const float* bo        = (const float*)d_in[9];
  const float* rel_bias  = (const float*)d_in[10];
  float* out = (float*)d_out;
  (void)in_sizes; (void)n_in; (void)out_size; (void)ws_size;

  const size_t nx  = (size_t)NBATCH * SEQ * HDIM;           // 4M
  const size_t nw  = (size_t)HDIM * HDIM;                   // 1M
  const size_t per = (size_t)NBATCH * NHEADS * SEQ * HEADD; // 4M

  _Float16* xh   = (_Float16*)d_ws;     // 8 MB
  _Float16* wh   = xh + nx;             // 8 MB (Wq|Wk|Wv|Wo, f16)
  _Float16* qh   = wh + 4 * nw;         // 8 MB
  _Float16* kh   = qh + per;            // 8 MB
  _Float16* vh   = kh + per;            // 8 MB (transposed [b,h,d,t])
  _Float16* hidh = vh + per;            // 8 MB

  cvt_f16_kernel<<<(int)(nx / 2048), 256, 0, stream>>>(x,  xh,          (int)nx);
  cvt_f16_kernel<<<(int)(nw / 2048), 256, 0, stream>>>(Wq, wh + 0 * nw, (int)nw);
  cvt_f16_kernel<<<(int)(nw / 2048), 256, 0, stream>>>(Wk, wh + 1 * nw, (int)nw);
  cvt_f16_kernel<<<(int)(nw / 2048), 256, 0, stream>>>(Wv, wh + 2 * nw, (int)nw);
  cvt_f16_kernel<<<(int)(nw / 2048), 256, 0, stream>>>(Wo, wh + 3 * nw, (int)nw);

  qkv_kernel    <<<1536, 256, 0, stream>>>(xh, wh, bq, bk, bv, qh, kh, vh);
  attn_kernel   <<<512,  256, 0, stream>>>(qh, kh, vh, text_mask, rel_bias, hidh);
  outproj_kernel<<<512,  256, 0, stream>>>(hidh, wh + 3 * nw, bo, text_mask, out);
}
